// FFN_43026982371490
// MI455X (gfx1250) — matrix operand order repaired and checked by reading, not compiled, not run on hardware
//
#include <hip/hip_runtime.h>
#include <hip/hip_bf16.h>

#define NTOK 8192   // B*S tokens
#define DIM  1024   // D
#define HID  2048   // H
#define NE   8      // experts
#define MT   64     // tokens per FFN block

#define SXPITCH 1032  // halves; 516 words = 4 mod 64 banks -> conflict-free A loads
#define SBPITCH 40    // halves; 20 words -> conflict-free B loads, 16B aligned
#define HBPITCH 40

typedef __attribute__((ext_vector_type(16))) __bf16 bf16x16;
typedef __attribute__((ext_vector_type(8)))  __bf16 bf16x8;
typedef __attribute__((ext_vector_type(8)))  float  f32x8;

static __device__ __forceinline__ unsigned short f32_to_bf16(float f) {
  unsigned int u = __builtin_bit_cast(unsigned int, f);
  u += 0x7fffu + ((u >> 16) & 1u);          // round-to-nearest-even
  return (unsigned short)(u >> 16);
}

// 32-bit LDS byte offset of a generic pointer into shared memory.
static __device__ __forceinline__ unsigned lds_off(const void* p) {
  return (unsigned)(unsigned long long)
         (__attribute__((address_space(3))) const void*)p;
}

// Async DMA: 16 bytes global -> LDS per lane, tracked by ASYNCcnt (CDNA5).
static __device__ __forceinline__ void async_copy16(unsigned lds, const void* g) {
  asm volatile("global_load_async_to_lds_b128 %0, %1, off"
               :: "v"(lds), "v"(g) : "memory");
}
static __device__ __forceinline__ void wait_async0() {
  asm volatile("s_wait_asynccnt 0" ::: "memory");
}

// WMMA A-operand (16x32 bf16) from 32 contiguous bf16 at rowbase.
// Lane m (0-15): K = {0..7, 16..23}; lane m+16: K = {8..15, 24..31}.
static __device__ __forceinline__ bf16x16 load_a32(const unsigned short* rowbase, int khalf) {
  bf16x8 lo = *(const bf16x8*)(rowbase + khalf);
  bf16x8 hi = *(const bf16x8*)(rowbase + 16 + khalf);
  return __builtin_shufflevector(lo, hi, 0,1,2,3,4,5,6,7,8,9,10,11,12,13,14,15);
}

// ---------------- Gate: logits -> softmax -> top2 -> expert lists ------------
__global__ __launch_bounds__(256)
void gate_kernel(const float* __restrict__ x, const float* __restrict__ gw,
                 const float* __restrict__ gb, unsigned short* __restrict__ xb,
                 float* __restrict__ out, int* __restrict__ counts,
                 int* __restrict__ etok, float* __restrict__ ecof)
{
  __shared__ float sgw[NE * DIM];                 // 32 KB: gate weights, read once
  int tid  = threadIdx.x;
  for (int i = tid; i < NE * DIM; i += 256) sgw[i] = gw[i];
  __syncthreads();

  int tok  = blockIdx.x * 8 + (tid >> 5);
  int lane = tid & 31;
  const float* xr = x + (size_t)tok * DIM;
  unsigned short* xbr = xb + (size_t)tok * DIM;
  float* outr = out + (size_t)tok * DIM;

  float acc[NE];
#pragma unroll
  for (int e = 0; e < NE; ++e) acc[e] = 0.f;

  for (int i = lane; i < DIM; i += 32) {
    float v = xr[i];
    xbr[i] = f32_to_bf16(v);
    outr[i] = 0.f;
#pragma unroll
    for (int e = 0; e < NE; ++e) acc[e] += v * sgw[e * DIM + i];
  }
#pragma unroll
  for (int e = 0; e < NE; ++e) {
#pragma unroll
    for (int off = 16; off > 0; off >>= 1)
      acc[e] += __shfl_xor(acc[e], off, 32);
  }
  if (lane == 0) {
    float l[NE], mx = -1e30f;
#pragma unroll
    for (int e = 0; e < NE; ++e) { l[e] = acc[e] + gb[e]; mx = fmaxf(mx, l[e]); }
    float p[NE];
#pragma unroll
    for (int e = 0; e < NE; ++e) p[e] = __expf(l[e] - mx);  // denom cancels in top2 renorm
    int i1 = 0;
#pragma unroll
    for (int e = 1; e < NE; ++e) if (p[e] > p[i1]) i1 = e;
    int i2 = (i1 == 0) ? 1 : 0;
#pragma unroll
    for (int e = 0; e < NE; ++e) if (e != i1 && p[e] > p[i2]) i2 = e;
    float denom = p[i1] + p[i2];
    float c1 = p[i1] / denom, c2 = p[i2] / denom;
    int s1 = atomicAdd(&counts[i1], 1);
    etok[i1 * NTOK + s1] = tok; ecof[i1 * NTOK + s1] = c1;
    int s2 = atomicAdd(&counts[i2], 1);
    etok[i2 * NTOK + s2] = tok; ecof[i2 * NTOK + s2] = c2;
  }
}

// ------------- Transpose + fp32->bf16 convert: src[R][C] -> dst[C][R] --------
__global__ __launch_bounds__(256)
void transpose_bf16(const float* __restrict__ src, unsigned short* __restrict__ dst,
                    int R, int C)
{
  __shared__ float t[32][33];
  size_t mb = (size_t)blockIdx.z * (size_t)R * (size_t)C;
  int c0 = blockIdx.x * 32, r0 = blockIdx.y * 32;
  int tx = threadIdx.x, ty = threadIdx.y;   // 32 x 8
#pragma unroll
  for (int i = 0; i < 32; i += 8)
    t[ty + i][tx] = src[mb + (size_t)(r0 + ty + i) * C + (c0 + tx)];
  __syncthreads();
#pragma unroll
  for (int i = 0; i < 32; i += 8)
    dst[mb + (size_t)(c0 + ty + i) * R + (r0 + tx)] = f32_to_bf16(t[tx][ty + i]);
}

// ---------------- Fused expert FFN -------------------------------------------
// Block: 64 gathered tokens of one expert, all 1024 output columns, 8 waves.
// W1/W3 per-kh slices double-pumped into LDS with async-to-LDS DMA (overlapped
// with h-formation and the second GEMM); wave w computes G1+G3 tile (mt=w>>1,
// nt=w&1) as two WMMA chains off one A; wave w owns out cols [w*128, w*128+128).
__global__ __launch_bounds__(256)
void moe_ffn(const unsigned short* __restrict__ xb,
             const unsigned short* __restrict__ w1t,   // [E][D][H] bf16
             const unsigned short* __restrict__ w3t,   // [E][D][H] bf16
             const unsigned short* __restrict__ w2t,   // [E][H][D] bf16
             const int* __restrict__ counts,
             const int* __restrict__ etok,
             const float* __restrict__ ecof,
             float* __restrict__ out)
{
  int e    = blockIdx.y;
  int cnt  = counts[e];
  int base = blockIdx.x * MT;
  if (base >= cnt) return;

  __shared__ __align__(16) unsigned short sx[MT * SXPITCH];     // 129 KB x tile
  __shared__ __align__(16) unsigned short sb1[DIM * SBPITCH];   // 80 KB W1 slice
  __shared__ __align__(16) unsigned short sb3[DIM * SBPITCH];   // 80 KB W3 slice
  __shared__ float gA[8][16][17];                               // 8.5 KB G1 tiles
  __shared__ float gB[8][16][17];                               // 8.5 KB G3 tiles
  __shared__ __align__(16) unsigned short hbuf[MT * HBPITCH];   // 5 KB h chunk
  __shared__ int   stok[MT];
  __shared__ float scof[MT];

  int tid  = threadIdx.x;
  int lane = tid & 31;
  int wave = tid >> 5;

  if (tid < MT) {
    int r = base + tid;
    stok[tid] = (r < cnt) ? etok[(size_t)e * NTOK + r] : 0;
    scof[tid] = (r < cnt) ? ecof[(size_t)e * NTOK + r] : 0.f;
  }
  __syncthreads();

  const unsigned short* w1e = w1t + (size_t)e * DIM * HID;
  const unsigned short* w3e = w3t + (size_t)e * DIM * HID;

  {  // async-stage gathered x tile (bf16): 64 rows x 1024, quarter-row/thread
    int row = tid & 63;
    int cb  = (tid >> 6) * 256;
    const unsigned short* src = xb + (size_t)stok[row] * DIM + cb;
    unsigned dstp = lds_off(sx + row * SXPITCH + cb);
#pragma unroll
    for (int i = 0; i < 256; i += 8)
      async_copy16(dstp + i * 2, src + i);
  }
  {  // async-stage W1/W3 slice for kh = 0
#pragma unroll
    for (int r = 0; r < 4; ++r) {
      int row = tid + r * 256;
      const unsigned short* s1 = w1e + (size_t)row * HID;
      const unsigned short* s3 = w3e + (size_t)row * HID;
      unsigned d1 = lds_off(sb1 + row * SBPITCH);
      unsigned d3 = lds_off(sb3 + row * SBPITCH);
#pragma unroll
      for (int c = 0; c < 32; c += 8) {
        async_copy16(d1 + c * 2, s1 + c);
        async_copy16(d3 + c * 2, s3 + c);
      }
    }
  }

  f32x8 acc[4][8];
  {
    f32x8 z = {0.f,0.f,0.f,0.f,0.f,0.f,0.f,0.f};
#pragma unroll
    for (int a = 0; a < 4; ++a)
#pragma unroll
      for (int b = 0; b < 8; ++b) acc[a][b] = z;
  }

  const int khalf = (lane < 16) ? 0 : 8;
  const int arow  = lane & 15;
  const int mt1   = wave >> 1;      // 0..3: step-1 token tile
  const int nt1   = wave & 1;       // 0..1: step-1 column tile

  wait_async0();
  __syncthreads();

  for (int kh = 0; kh < HID; kh += 32) {
    // ---- step 1: G1/G3 [64 x 32]; wave w -> tile (mt1, nt1), 2 WMMA chains --
    {
      f32x8 g1 = {0.f,0.f,0.f,0.f,0.f,0.f,0.f,0.f};
      f32x8 g3 = g1;
      const unsigned short* sa = sx  + (mt1 * 16 + arow) * SXPITCH;
      const unsigned short* b1 = sb1 + nt1 * 16 + arow;
      const unsigned short* b3 = sb3 + nt1 * 16 + arow;
#pragma unroll 4
      for (int k0 = 0; k0 < DIM; k0 += 32) {
        bf16x16 a   = load_a32(sa + k0, khalf);
        bf16x16 bb1, bb3;
#pragma unroll
        for (int s = 0; s < 16; ++s) {
          int kk = k0 + khalf + (s & 7) + ((s >> 3) << 4);
          bb1[s] = __builtin_bit_cast(__bf16, b1[kk * SBPITCH]);
          bb3[s] = __builtin_bit_cast(__bf16, b3[kk * SBPITCH]);
        }
        g1 = __builtin_amdgcn_wmma_f32_16x16x32_bf16(false, a, false, bb1,
                                                     (short)0, g1, false, false);
        g3 = __builtin_amdgcn_wmma_f32_16x16x32_bf16(false, a, false, bb3,
                                                     (short)0, g3, false, false);
      }
      int rb = (lane < 16) ? 0 : 8;
#pragma unroll
      for (int r = 0; r < 8; ++r) {
        gA[wave][rb + r][arow] = g1[r];
        gB[wave][rb + r][arow] = g3[r];
      }
    }
    __syncthreads();   // gbuf ready; sb(kh) fully consumed

    // ---- overlap: async-stage W1/W3 slice for kh+32 while we form h ----
    if (kh + 32 < HID) {
#pragma unroll
      for (int r = 0; r < 4; ++r) {
        int row = tid + r * 256;
        const unsigned short* s1 = w1e + (size_t)row * HID + kh + 32;
        const unsigned short* s3 = w3e + (size_t)row * HID + kh + 32;
        unsigned d1 = lds_off(sb1 + row * SBPITCH);
        unsigned d3 = lds_off(sb3 + row * SBPITCH);
#pragma unroll
        for (int c = 0; c < 32; c += 8) {
          async_copy16(d1 + c * 2, s1 + c);
          async_copy16(d3 + c * 2, s3 + c);
        }
      }
    }

    // ---- prefetch W2 B-tiles for this kh (consumed in step 2b) ----
    bf16x16 bpre[8];
    {
      const unsigned short* wb =
          w2t + ((size_t)e * HID + kh) * DIM + wave * 128 + arow;
#pragma unroll
      for (int nt = 0; nt < 8; ++nt)
#pragma unroll
        for (int s = 0; s < 16; ++s)
          bpre[nt][s] = __builtin_bit_cast(__bf16,
              wb[(size_t)(khalf + (s & 7) + ((s >> 3) << 4)) * DIM + nt * 16]);
    }

    // ---- step 2a: h = relu(G1)^2 * G3 -> bf16 in LDS (8 elems/thread) ----
#pragma unroll
    for (int i = 0; i < 8; ++i) {
      int idx = tid + i * 256;
      int m = idx >> 5, j = idx & 31;
      int w = ((m >> 4) << 1) | (j >> 4);
      float g1 = gA[w][m & 15][j & 15];
      float g3 = gB[w][m & 15][j & 15];
      float r  = g1 > 0.f ? g1 : 0.f;
      hbuf[m * HBPITCH + j] = f32_to_bf16(r * r * g3);
    }
    __syncthreads();

    // ---- step 2b: out[64, wave*128..+128) += h[64 x 32] * W2t chunk ----
#pragma unroll
    for (int mt = 0; mt < 4; ++mt) {
      bf16x16 a = load_a32(hbuf + (mt * 16 + arow) * HBPITCH, khalf);
#pragma unroll
      for (int nt = 0; nt < 8; ++nt)
        acc[mt][nt] = __builtin_amdgcn_wmma_f32_16x16x32_bf16(
            false, a, false, bpre[nt], (short)0, acc[mt][nt], false, false);
    }

    wait_async0();     // sb(kh+32) resident in LDS
    __syncthreads();
  }

  // ---- epilogue: scale by routing coef, scatter-add into out ----
  {
    int rb = (lane < 16) ? 0 : 8;
#pragma unroll
    for (int mt = 0; mt < 4; ++mt)
#pragma unroll
      for (int r = 0; r < 8; ++r) {
        int row = mt * 16 + rb + r;
        float cf = scof[row];
        float* op = out + (size_t)stok[row] * DIM + wave * 128 + arow;
#pragma unroll
        for (int nt = 0; nt < 8; ++nt)
          atomicAdd(op + nt * 16, acc[mt][nt][r] * cf);
      }
  }
}

extern "C" void kernel_launch(void* const* d_in, const int* in_sizes, int n_in,
                              void* d_out, int out_size, void* d_ws, size_t ws_size,
                              hipStream_t stream) {
  (void)in_sizes; (void)n_in; (void)out_size; (void)ws_size;
  const float* x  = (const float*)d_in[0];
  const float* W1 = (const float*)d_in[1];
  const float* W2 = (const float*)d_in[2];
  const float* W3 = (const float*)d_in[3];
  const float* gw = (const float*)d_in[4];
  const float* gb = (const float*)d_in[5];
  float* out = (float*)d_out;

  char* wsb = (char*)d_ws;
  size_t off = 0;
  auto take = [&](size_t bytes) -> char* {
    char* p = wsb + off;
    off += (bytes + 255) & ~(size_t)255;
    return p;
  };
  int*            counts = (int*)take(NE * sizeof(int));
  int*            etok   = (int*)take((size_t)NE * NTOK * sizeof(int));
  float*          ecof   = (float*)take((size_t)NE * NTOK * sizeof(float));
  unsigned short* xb     = (unsigned short*)take((size_t)NTOK * DIM * 2);
  unsigned short* w1t    = (unsigned short*)take((size_t)NE * DIM * HID * 2);
  unsigned short* w3t    = (unsigned short*)take((size_t)NE * DIM * HID * 2);
  unsigned short* w2t    = (unsigned short*)take((size_t)NE * HID * DIM * 2);

  hipMemsetAsync(counts, 0, NE * sizeof(int), stream);

  gate_kernel<<<NTOK / 8, 256, 0, stream>>>(x, gw, gb, xb, out, counts, etok, ecof);

  dim3 tb(32, 8);
  transpose_bf16<<<dim3(DIM / 32, HID / 32, NE), tb, 0, stream>>>(W1, w1t, HID, DIM);
  transpose_bf16<<<dim3(DIM / 32, HID / 32, NE), tb, 0, stream>>>(W3, w3t, HID, DIM);
  transpose_bf16<<<dim3(HID / 32, DIM / 32, NE), tb, 0, stream>>>(W2, w2t, DIM, HID);

  moe_ffn<<<dim3(NTOK / MT, NE), 256, 0, stream>>>(xb, w1t, w3t, w2t,
                                                   counts, etok, ecof, out);
}
